// Attention_concat_65231963291707
// MI455X (gfx1250) — compile-verified, matrix operand order checked by reading, not run on hardware
//
#include <hip/hip_runtime.h>
#include <math.h>
#include <stdint.h>

// ---------------------------------------------------------------------------
// Additive attention, fused for MI455X (gfx1250, wave32, WMMA).
// Dominant GEMM (131072x512 @ 512x512, 68.7 GFLOP) runs on
// v_wmma_f32_16x16x32_bf16 with context converted f32->bf16 on the fly into
// LDS A-fragments, and the weight block pre-swizzled once into B-fragment
// layout (512 KB, L2-resident). Four N-tile accumulators are kept live so
// each A-fragment LDS read feeds 4 WMMAs. tanh + v_w contraction fused in
// registers (native v_tanh_f32) so the (B,T,D) energy tensor is never
// written to memory.
// ---------------------------------------------------------------------------

#define B_ 32
#define T_ 4096
#define D_ 512

typedef __bf16 bf16_t;
typedef bf16_t v16bf __attribute__((ext_vector_type(16)));
typedef float  v8f   __attribute__((ext_vector_type(8)));

__device__ __forceinline__ uint32_t pack_bf16(float a, float b) {
    uint32_t ua = __float_as_uint(a), ub = __float_as_uint(b);
    uint32_t ra = (ua + 0x7FFFu + ((ua >> 16) & 1u)) >> 16;   // RNE
    uint32_t rb = (ub + 0x7FFFu + ((ub >> 16) & 1u)) >> 16;
    return (ra & 0xFFFFu) | (rb << 16);
}

// Native CDNA5 v_tanh_f32 (TRANS32 op).
__device__ __forceinline__ float fast_tanh(float x) {
#if __has_builtin(__builtin_amdgcn_tanhf)
    return __builtin_amdgcn_tanhf(x);
#else
    float r;
    asm volatile("v_tanh_f32 %0, %1\n\tv_nop" : "=v"(r) : "v"(x));
    return r;
#endif
}

// ---------------------------------------------------------------------------
// q[b,h] = attn_b[h] + sum_k output[b,k] * attn_w[h,k]   (W1 half of attn_w)
// grid (32,2) x 256
__global__ void prep_q_kernel(const float* __restrict__ output,
                              const float* __restrict__ attn_w,
                              const float* __restrict__ attn_b,
                              float* __restrict__ q) {
    int b = blockIdx.x;
    int h = blockIdx.y * 256 + threadIdx.x;
    const float* o = output + (size_t)b * D_;
    const float* w = attn_w + (size_t)h * (2 * D_);
    float acc = attn_b[h];
    for (int k = 0; k < D_; ++k) acc += o[k] * w[k];
    q[b * D_ + h] = acc;
}

// ---------------------------------------------------------------------------
// Swizzle W2 = attn_w[:, D:].T (shape K=512 x N=512) into bf16 WMMA
// B-fragment layout: dword index = ((nt*16 + ks)*32 + lane)*8 + d
//   lane n-half: n = nt*16 + (lane&15); k-offset half = (lane>>4)*8
//   d<4 -> kk2 = 2d ; d>=4 -> kk2 = 16 + 2(d-4);  k = ks*32 + kk2 + 8*(lane>>4)
// grid 512 x 256 (131072 dwords total)
__global__ void prep_w2_kernel(const float* __restrict__ attn_w,
                               uint32_t* __restrict__ w2s) {
    int idx = blockIdx.x * 256 + threadIdx.x;
    int d   = idx & 7;
    int l   = (idx >> 3) & 31;
    int ks  = (idx >> 8) & 15;
    int nt  = idx >> 12;
    int n   = nt * 16 + (l & 15);
    int kk2 = (d < 4) ? (2 * d) : (16 + 2 * (d - 4));
    int k   = ks * 32 + kk2 + 8 * (l >> 4);
    const float* row = attn_w + (size_t)n * (2 * D_) + D_;  // W2 half
    w2s[idx] = pack_bf16(row[k], row[k + 1]);
}

// ---------------------------------------------------------------------------
// Fused scores kernel: per block 128 context rows; per wave one 16-row tile.
// scores[b,t] = v_b + sum_h v_w[h] * tanh(q[b,h] + context[b,t]·W2[h])
// grid (32, 32) x 256, dynamic LDS = 128 KB (A-fragments, bf16)
__global__ void score_kernel(const float* __restrict__ ctx,
                             const float* __restrict__ q,
                             const float* __restrict__ vw,
                             const float* __restrict__ vb,
                             const uint32_t* __restrict__ w2s,
                             float* __restrict__ scores) {
    extern __shared__ uint32_t smem[];   // [mt(8)][ks(16)][lane(32)][d(8)] dwords
    const int b    = blockIdx.x;
    const int t0   = blockIdx.y * 128;
    const int tid  = threadIdx.x;
    const int lane = tid & 31;
    const int mt   = tid >> 5;           // wave id == M-tile id

    // ---- stage 128 rows of context as bf16 A-fragments --------------------
    const float* cbase = ctx + ((size_t)b * T_ + t0) * D_;
    for (int idx = tid; idx < 128 * 256; idx += 256) {
        int m  = idx >> 8;               // row 0..127
        int k  = (idx & 255) * 2;        // even k
        float2 f = *reinterpret_cast<const float2*>(cbase + (size_t)m * D_ + k);
        uint32_t pk = pack_bf16(f.x, f.y);
        int kk  = k & 31;
        int ks  = k >> 5;
        int lh  = (kk >> 3) & 1;         // which lane-half holds this k
        int kk2 = kk - 8 * lh;
        int d   = (kk2 < 8) ? (kk2 >> 1) : (4 + ((kk2 - 16) >> 1));
        int l   = (m & 15) + (lh << 4);
        smem[((((m >> 4) * 16 + ks) * 32 + l) << 3) + d] = pk;
    }
    __syncthreads();

    // ---- WMMA GEMM + fused tanh / v_w contraction -------------------------
    const v16bf* ap = reinterpret_cast<const v16bf*>(smem) + (mt * 16) * 32 + lane;
    const float* qb = q + b * D_;

    float sp[8];
#pragma unroll
    for (int r = 0; r < 8; ++r) sp[r] = 0.0f;

    // 8 groups of 4 N-tiles; A-fragment read once per ks, feeds 4 WMMAs.
    for (int ng = 0; ng < 8; ++ng) {
        const v16bf* wp = reinterpret_cast<const v16bf*>(w2s)
                        + (ng * 4 * 16) * 32 + lane;
        v8f acc[4];
#pragma unroll
        for (int j = 0; j < 4; ++j)
            acc[j] = (v8f){0.f, 0.f, 0.f, 0.f, 0.f, 0.f, 0.f, 0.f};
#pragma unroll 2
        for (int ks = 0; ks < 16; ++ks) {
            v16bf a = ap[ks * 32];
#pragma unroll
            for (int j = 0; j < 4; ++j) {
                v16bf bf = wp[(j * 16 + ks) * 32];
                acc[j] = __builtin_amdgcn_wmma_f32_16x16x32_bf16(
                    false, a, false, bf, (short)0, acc[j], false, false);
            }
        }
        // acc VGPR r, lane<16: E[m=r, n=lane]; lane>=16: E[m=8+r, n=lane-16]
#pragma unroll
        for (int j = 0; j < 4; ++j) {
            int   h   = (ng * 4 + j) * 16 + (lane & 15);
            float qv  = qb[h];
            float vwv = vw[h];
#pragma unroll
            for (int r = 0; r < 8; ++r) sp[r] += fast_tanh(acc[j][r] + qv) * vwv;
        }
    }

    // reduce over the 16 lanes of each half-wave (N dimension)
#pragma unroll
    for (int off = 1; off < 16; off <<= 1) {
#pragma unroll
        for (int r = 0; r < 8; ++r) sp[r] += __shfl_xor(sp[r], off, 32);
    }

    float vbv = vb[0];
    if ((lane & 15) == 0) {
        int m0 = mt * 16 + ((lane >> 4) << 3);   // lane0 -> rows +0..7, lane16 -> +8..15
        float* srow = scores + (size_t)b * T_ + t0 + m0;
#pragma unroll
        for (int r = 0; r < 8; ++r) srow[r] = sp[r] + vbv;
    }
}

// ---------------------------------------------------------------------------
// Softmax over T per batch. grid 32 x 256.
__global__ void softmax_kernel(const float* __restrict__ scores,
                               float* __restrict__ attn) {
    __shared__ float red[256];
    int b = blockIdx.x, tid = threadIdx.x;
    const float* s = scores + (size_t)b * T_;
    float mx = -3.4e38f;
    for (int t = tid; t < T_; t += 256) mx = fmaxf(mx, s[t]);
    red[tid] = mx; __syncthreads();
    for (int st = 128; st > 0; st >>= 1) {
        if (tid < st) red[tid] = fmaxf(red[tid], red[tid + st]);
        __syncthreads();
    }
    mx = red[0]; __syncthreads();
    float sum = 0.0f;
    for (int t = tid; t < T_; t += 256) sum += __expf(s[t] - mx);
    red[tid] = sum; __syncthreads();
    for (int st = 128; st > 0; st >>= 1) {
        if (tid < st) red[tid] += red[tid + st];
        __syncthreads();
    }
    float inv = 1.0f / red[0];
    float* a = attn + (size_t)b * T_;
    for (int t = tid; t < T_; t += 256) a[t] = __expf(s[t] - mx) * inv;
}

// ---------------------------------------------------------------------------
// mix partials: mixp[tc][b*512+d] = sum_{t in chunk} attn[b,t]*ctx[b,t,d]
// grid (32, 2, 8) x 256 — deterministic (no float atomics).
__global__ void mix_partial_kernel(const float* __restrict__ attn,
                                   const float* __restrict__ ctx,
                                   float* __restrict__ mixp) {
    int b  = blockIdx.x;
    int d  = blockIdx.y * 256 + threadIdx.x;
    int tc = blockIdx.z;
    const float* a = attn + (size_t)b * T_;
    const float* c = ctx + ((size_t)b * T_) * D_ + d;
    float acc = 0.0f;
    int t0 = tc * (T_ / 8);
    for (int t = t0; t < t0 + T_ / 8; ++t) acc += a[t] * c[(size_t)t * D_];
    mixp[(size_t)tc * (B_ * D_) + b * D_ + d] = acc;
}

// grid 64 x 256
__global__ void mix_reduce_kernel(const float* __restrict__ mixp,
                                  float* __restrict__ mixws) {
    int i = blockIdx.x * 256 + threadIdx.x;
    float s = 0.0f;
    for (int p = 0; p < 8; ++p) s += mixp[(size_t)p * (B_ * D_) + i];
    mixws[i] = s;
}

// ---------------------------------------------------------------------------
// out[b,h] = tanh(out_b[h] + mix[b]·out_w[h,:D] + output[b]·out_w[h,D:])
// also emits mix to d_out.  grid (32,2) x 256
__global__ void final_kernel(const float* __restrict__ mixws,
                             const float* __restrict__ output,
                             const float* __restrict__ out_w,
                             const float* __restrict__ out_b,
                             float* __restrict__ out,
                             float* __restrict__ mix_out) {
    int b = blockIdx.x;
    int h = blockIdx.y * 256 + threadIdx.x;
    const float* w = out_w + (size_t)h * (2 * D_);
    const float* m = mixws + b * D_;
    const float* o = output + b * D_;
    float acc = out_b[h];
    for (int k = 0; k < D_; ++k) acc += m[k] * w[k];
    for (int k = 0; k < D_; ++k) acc += o[k] * w[D_ + k];
    out[b * D_ + h]     = fast_tanh(acc);
    mix_out[b * D_ + h] = m[h];
}

// ---------------------------------------------------------------------------
extern "C" void kernel_launch(void* const* d_in, const int* in_sizes, int n_in,
                              void* d_out, int out_size, void* d_ws, size_t ws_size,
                              hipStream_t stream) {
    (void)in_sizes; (void)n_in; (void)out_size; (void)ws_size;
    const float* output = (const float*)d_in[0];  // (B,1,D)
    const float* ctx    = (const float*)d_in[1];  // (B,T,D)
    const float* attn_w = (const float*)d_in[2];  // (D,2D)
    const float* attn_b = (const float*)d_in[3];  // (D,)
    const float* v_w    = (const float*)d_in[4];  // (1,D)
    const float* v_b    = (const float*)d_in[5];  // (1,)
    const float* out_w  = (const float*)d_in[6];  // (D,2D)
    const float* out_b  = (const float*)d_in[7];  // (D,)

    // d_out layout: out (B*D) | attn (B*T) | mix (B*D)
    float* out_p  = (float*)d_out;
    float* attn_p = out_p + B_ * D_;
    float* mix_p  = attn_p + B_ * T_;

    // workspace layout (bytes)
    char* ws = (char*)d_ws;
    float*    q_ws    = (float*)(ws);                       //  64 KB
    uint32_t* w2s_ws  = (uint32_t*)(ws + 65536);            // 512 KB bf16 B-frags
    float*    sc_ws   = (float*)(ws + 65536 + 524288);      // 512 KB scores
    float*    mixp_ws = (float*)(ws + 65536 + 2 * 524288);  // 512 KB partials
    float*    mix_ws  = (float*)(ws + 65536 + 3 * 524288);  //  64 KB mix

    prep_q_kernel<<<dim3(B_, 2), 256, 0, stream>>>(output, attn_w, attn_b, q_ws);
    prep_w2_kernel<<<512, 256, 0, stream>>>(attn_w, w2s_ws);
    score_kernel<<<dim3(B_, T_ / 128), 256, 131072, stream>>>(
        ctx, q_ws, v_w, v_b, w2s_ws, sc_ws);
    softmax_kernel<<<B_, 256, 0, stream>>>(sc_ws, attn_p);
    mix_partial_kernel<<<dim3(B_, 2, 8), 256, 0, stream>>>(attn_p, ctx, mixp_ws);
    mix_reduce_kernel<<<64, 256, 0, stream>>>(mixp_ws, mix_ws);
    final_kernel<<<dim3(B_, 2), 256, 0, stream>>>(mix_ws, output, out_w, out_b,
                                                  out_p, mix_p);
}
